// ECC_Transformer_23837068493260
// MI455X (gfx1250) — compile-verified, matrix operand order checked by reading, not run on hardware
//
#include <hip/hip_runtime.h>
#include <hip/hip_bf16.h>
#include <math.h>

typedef __attribute__((ext_vector_type(16))) _Float16 v16h;
typedef __attribute__((ext_vector_type(8)))  float    v8f;
typedef __attribute__((ext_vector_type(8)))  _Float16 h8;

#define BS    32
#define SEQL  1536
#define DIM   256
#define NH    8
#define DHD   32
#define MF    110
#define MP    112
#define NROWS (BS * SEQL)

// Performer constants: nrm = 32^-0.25, diag scale = 0.5*nrm^2, ratio = 110^-0.5
#define C_NRM   0.42044820762685725f
#define C_DIAG  0.08838834764831845f
#define C_RATIO 0.09534625892455924f
#define C_EPS   1e-4f

// ---- monotone float<->uint encoding for atomic max over signed floats ----
__device__ __forceinline__ unsigned enc_f(float f) {
  unsigned u = __float_as_uint(f);
  return (u & 0x80000000u) ? ~u : (u | 0x80000000u);
}
__device__ __forceinline__ float dec_f(unsigned e) {
  unsigned u = (e & 0x80000000u) ? (e ^ 0x80000000u) : ~e;
  return __uint_as_float(u);
}

// ---- CDNA5 async global->LDS copy (16B per lane), ASYNCcnt tracked ----
__device__ __forceinline__ void async_copy16(_Float16* lds_dst, const _Float16* gsrc) {
  unsigned lo = (unsigned)(uintptr_t)lds_dst;           // LDS byte offset (low 32 bits of generic)
  unsigned long long ga = (unsigned long long)(uintptr_t)gsrc;
  asm volatile("global_load_async_to_lds_b128 %0, %1, off"
               :: "v"(lo), "v"(ga) : "memory");
}
template<int N>
__device__ __forceinline__ void wait_async() {
#if __has_builtin(__builtin_amdgcn_s_wait_asynccnt)
  __builtin_amdgcn_s_wait_asynccnt(N);
#else
  asm volatile("s_wait_asynccnt %0" :: "i"(N) : "memory");
#endif
}

// ============================================================================
// WMMA GEMM: C[rows,N] = A[rows,K] @ W[K,N] + bias (+ epilogue)
//   A:  f16 row-major [rows][K]
//   Wt: f16 pre-transposed weights [N][K]
// EPI: 0 = bias, 1 = bias+GELU(exact), 2 = bias+residual(f32)
// OUT16: 1 -> f16 output, 0 -> f32 output
// Block 128 threads (4 waves), tile 64x64, K-step 32, f16 WMMA / f32 acc.
// Tiles double-buffered in LDS via global_load_async_to_lds_b128; compute on
// buffer `cur` overlaps the async fill of buffer `cur^1` (wait asynccnt<=4).
// ============================================================================
template<int EPI, int OUT16>
__global__ __launch_bounds__(128)
void gemm_f16_kernel(const _Float16* __restrict__ A, const _Float16* __restrict__ Wt,
                     const float* __restrict__ bias, const float* __restrict__ res,
                     void* __restrict__ Cv, int K, int N) {
  __shared__ _Float16 As[2][64][40];   // [buf][m][k], +8 halves pad (16B) per row
  __shared__ _Float16 Bt[2][64][40];   // [buf][n][k]
  const int tid  = threadIdx.x;
  const int lane = tid & 31;
  const int wave = tid >> 5;
  const int rowBase = blockIdx.y << 6;
  const int colBase = blockIdx.x << 6;
  const int wm = (wave >> 1) << 5;
  const int wn = (wave & 1) << 5;

  v8f acc[2][2];
#pragma unroll
  for (int i = 0; i < 2; ++i)
#pragma unroll
    for (int j = 0; j < 2; ++j)
#pragma unroll
      for (int r = 0; r < 8; ++r) acc[i][j][r] = 0.f;

  // stage a 64x32 f16 tile pair: 4 async b128 instructions per wave
  auto stage = [&](int buf, int k0) {
#pragma unroll
    for (int i = 0; i < 2; ++i) {
      int idx = tid + (i << 7);        // 0..255
      int r   = idx >> 2;              // 0..63
      int c8  = (idx & 3) << 3;        // 0,8,16,24 halves
      async_copy16(&As[buf][r][c8], A  + (size_t)(rowBase + r) * K + k0 + c8);
      async_copy16(&Bt[buf][r][c8], Wt + (size_t)(colBase + r) * K + k0 + c8);
    }
  };

  stage(0, 0);
  const int steps = K >> 5;
  for (int s = 0; s < steps; ++s) {
    const int cur = s & 1;
    if (s + 1 < steps) {
      stage(cur ^ 1, (s + 1) << 5);   // prefetch next tile into other buffer
      wait_async<4>();                // current tile's (older) copies complete
    } else {
      wait_async<0>();
    }
    __syncthreads();

    // fragment loads per ISA 16-bit A 16x32 layout (B mirrored via transpose)
    const int fr = lane & 15;
    const int kb = (lane >> 4) << 3;   // 0 or 8
    v16h af[2], bf[2];
#pragma unroll
    for (int i = 0; i < 2; ++i) {
      const _Float16* pa = &As[cur][wm + (i << 4) + fr][0];
      h8 alo = *(const h8*)(pa + kb);
      h8 ahi = *(const h8*)(pa + kb + 16);
      const _Float16* pb = &Bt[cur][wn + (i << 4) + fr][0];
      h8 blo = *(const h8*)(pb + kb);
      h8 bhi = *(const h8*)(pb + kb + 16);
#pragma unroll
      for (int j = 0; j < 8; ++j) {
        af[i][j] = alo[j]; af[i][j + 8] = ahi[j];
        bf[i][j] = blo[j]; bf[i][j + 8] = bhi[j];
      }
    }
#pragma unroll
    for (int i = 0; i < 2; ++i)
#pragma unroll
      for (int j = 0; j < 2; ++j)
        acc[i][j] = __builtin_amdgcn_wmma_f32_16x16x32_f16(
            false, af[i], false, bf[j], (short)0, acc[i][j], false, false);
    __syncthreads();
  }

  // epilogue: C/D layout — lane holds N = lane%16, VGPR r holds M = r + 8*(lane/16)
  const int nc = lane & 15;
  const int mr = (lane >> 4) << 3;
#pragma unroll
  for (int i = 0; i < 2; ++i) {
#pragma unroll
    for (int j = 0; j < 2; ++j) {
      int col = colBase + wn + (j << 4) + nc;
      float bb = bias[col];
#pragma unroll
      for (int r = 0; r < 8; ++r) {
        int row = rowBase + wm + (i << 4) + mr + r;
        float val = acc[i][j][r] + bb;
        if (EPI == 1) val = 0.5f * val * (1.0f + erff(val * 0.7071067811865475f));
        if (EPI == 2) val += res[(size_t)row * N + col];
        if (OUT16) ((_Float16*)Cv)[(size_t)row * N + col] = (_Float16)val;
        else       ((float*)Cv)[(size_t)row * N + col] = val;
      }
    }
  }
}

// ============================================================================
// Weight convert+transpose: W [count][K][N] f32 -> Wt [count][N][K] f16
// ============================================================================
__global__ void convT_kernel(const float* __restrict__ W, _Float16* __restrict__ Wt,
                             int K, int N, int count) {
  const size_t per = (size_t)K * N;
  const size_t total = per * count;
  for (size_t idx = (size_t)blockIdx.x * blockDim.x + threadIdx.x; idx < total;
       idx += (size_t)gridDim.x * blockDim.x) {
    size_t m = idx / per, r = idx % per;
    int k = (int)(r / N), n = (int)(r % N);
    Wt[m * per + (size_t)n * K + k] = (_Float16)W[idx];
  }
}

// ============================================================================
// LayerNorm over D=256, one block per row -> f16 output (GEMM operand)
// ============================================================================
__global__ __launch_bounds__(256)
void ln_kernel(const float* __restrict__ X, const float* __restrict__ S,
               const float* __restrict__ B, _Float16* __restrict__ Y) {
  __shared__ float red[256];
  const int tid = threadIdx.x;
  const size_t base = (size_t)blockIdx.x * DIM;
  float v = X[base + tid];
  red[tid] = v;
  __syncthreads();
  for (int s = 128; s > 0; s >>= 1) {
    if (tid < s) red[tid] += red[tid + s];
    __syncthreads();
  }
  float mu = red[0] * (1.0f / DIM);
  __syncthreads();
  float dv = v - mu;
  red[tid] = dv * dv;
  __syncthreads();
  for (int s = 128; s > 0; s >>= 1) {
    if (tid < s) red[tid] += red[tid + s];
    __syncthreads();
  }
  float var = red[0] * (1.0f / DIM);
  float r = rsqrtf(var + 1e-5f);
  Y[base + tid] = (_Float16)(dv * r * S[tid] + B[tid]);
}

// ============================================================================
// Performer feature maps. Q/K stored [row=b*SEQ+n][256] f32, head slice h*32.
// qp/kp layout: [b,h,n,m] with m stride MP=112 (pad zeros).
// ============================================================================
__global__ __launch_bounds__(128)
void phi_q_kernel(const float* __restrict__ Q, const float* __restrict__ proj,
                  float* __restrict__ QP) {
  __shared__ float qs[DHD];
  __shared__ float red[128];
  const int tid = threadIdx.x;
  const int gid = blockIdx.x;              // ((b*NH+h)*SEQL + n)
  const int n  = gid % SEQL;
  const int bh = gid / SEQL;
  const int hh = bh % NH;
  const int bb = bh / NH;
  const float* qrow = Q + ((size_t)(bb * SEQL + n)) * DIM + hh * DHD;
  if (tid < DHD) qs[tid] = qrow[tid];
  __syncthreads();
  float qq = 0.f;
#pragma unroll
  for (int d = 0; d < DHD; ++d) qq += qs[d] * qs[d];
  const float diag = qq * C_DIAG;
  float myxd = -3.0e38f;
  if (tid < MF) {
    const float* pr = proj + tid * DHD;
    float s = 0.f;
#pragma unroll
    for (int d = 0; d < DHD; ++d) s += qs[d] * pr[d];
    myxd = s * C_NRM;
  }
  red[tid] = myxd;
  __syncthreads();
  for (int s = 64; s > 0; s >>= 1) {
    if (tid < s) red[tid] = fmaxf(red[tid], red[tid + s]);
    __syncthreads();
  }
  const float stab = red[0];
  float* outp = QP + (size_t)gid * MP;
  if (tid < MP)
    outp[tid] = (tid < MF) ? C_RATIO * (expf(myxd - diag - stab) + C_EPS) : 0.f;
}

__global__ __launch_bounds__(128)
void phi_k1_kernel(const float* __restrict__ Kt, const float* __restrict__ proj,
                   float* __restrict__ KX, unsigned* __restrict__ kmax) {
  __shared__ float qs[DHD];
  __shared__ float red[128];
  const int tid = threadIdx.x;
  const int gid = blockIdx.x;
  const int n  = gid % SEQL;
  const int bh = gid / SEQL;
  const int hh = bh % NH;
  const int bb = bh / NH;
  const float* krow = Kt + ((size_t)(bb * SEQL + n)) * DIM + hh * DHD;
  if (tid < DHD) qs[tid] = krow[tid];
  __syncthreads();
  float qq = 0.f;
#pragma unroll
  for (int d = 0; d < DHD; ++d) qq += qs[d] * qs[d];
  const float diag = qq * C_DIAG;
  float myxd = -3.0e38f;
  if (tid < MF) {
    const float* pr = proj + tid * DHD;
    float s = 0.f;
#pragma unroll
    for (int d = 0; d < DHD; ++d) s += qs[d] * pr[d];
    myxd = s * C_NRM;
  }
  if (tid < MP) KX[(size_t)gid * MP + tid] = (tid < MF) ? (myxd - diag) : 0.f;
  red[tid] = myxd;
  __syncthreads();
  for (int s = 64; s > 0; s >>= 1) {
    if (tid < s) red[tid] = fmaxf(red[tid], red[tid + s]);
    __syncthreads();
  }
  if (tid == 0) atomicMax(&kmax[bh], enc_f(red[0]));
}

__global__ void phi_k2_kernel(float* __restrict__ KX, const unsigned* __restrict__ kmax) {
  const size_t total = (size_t)BS * NH * SEQL * MP;
  for (size_t idx = (size_t)blockIdx.x * blockDim.x + threadIdx.x; idx < total;
       idx += (size_t)gridDim.x * blockDim.x) {
    int m = (int)(idx % MP);
    size_t row = idx / MP;
    int bh = (int)(row / SEQL);
    float stab = dec_f(kmax[bh]);
    float t = KX[idx];
    KX[idx] = (m < MF) ? C_RATIO * (expf(t - stab) + C_EPS) : 0.f;
  }
}

__global__ __launch_bounds__(128)
void ksum_kernel(const float* __restrict__ KP, float* __restrict__ KS) {
  const int bh = blockIdx.x;
  const int tid = threadIdx.x;
  if (tid >= MP) return;
  const float* kb = KP + (size_t)bh * SEQL * MP + tid;
  float s = 0.f;
  for (int n = 0; n < SEQL; ++n) s += kb[(size_t)n * MP];
  KS[bh * MP + tid] = s;
}

// ctx[b,h,m,d] = sum_n kp[b,h,n,m] * v[b,h,n,d]
__global__ __launch_bounds__(128)
void ctx_kernel(const float* __restrict__ KP, const float* __restrict__ V,
                float* __restrict__ CTX) {
  __shared__ float vs[DHD];
  const int bh = blockIdx.x;
  const int tid = threadIdx.x;
  const int bb = bh / NH, hh = bh % NH;
  const float* kpb = KP + (size_t)bh * SEQL * MP;
  const float* vb  = V + (size_t)bb * SEQL * DIM + hh * DHD;
  float acc[DHD];
#pragma unroll
  for (int d = 0; d < DHD; ++d) acc[d] = 0.f;
  for (int n = 0; n < SEQL; ++n) {
    if (tid < DHD) vs[tid] = vb[(size_t)n * DIM + tid];
    __syncthreads();
    float kv = (tid < MP) ? kpb[(size_t)n * MP + tid] : 0.f;
#pragma unroll
    for (int d = 0; d < DHD; ++d) acc[d] += kv * vs[d];
    __syncthreads();
  }
  if (tid < MP) {
    float* outp = CTX + ((size_t)bh * MP + tid) * DHD;
#pragma unroll
    for (int d = 0; d < DHD; ++d) outp[d] = acc[d];
  }
}

// o[b,h,n,d] = (qp[n,:] . ctx[:,d]) / (qp[n,:] . ksum)  -> f16 (GEMM operand)
__global__ __launch_bounds__(128)
void attn_o_kernel(const float* __restrict__ QP, const float* __restrict__ CTX,
                   const float* __restrict__ KS, _Float16* __restrict__ O) {
  __shared__ float qs[MP];
  __shared__ float red[128];
  const int tid = threadIdx.x;
  const int gid = blockIdx.x;
  const int n  = gid % SEQL;
  const int bh = gid / SEQL;
  const float* qpr = QP + (size_t)gid * MP;
  float qv = (tid < MP) ? qpr[tid] : 0.f;
  if (tid < MP) qs[tid] = qv;
  red[tid] = (tid < MP) ? qv * KS[bh * MP + tid] : 0.f;
  __syncthreads();
  for (int s = 64; s > 0; s >>= 1) {
    if (tid < s) red[tid] += red[tid + s];
    __syncthreads();
  }
  const float dinv = 1.0f / red[0];
  if (tid < DHD) {
    const float* ctxb = CTX + (size_t)bh * MP * DHD;
    float s = 0.f;
    for (int m = 0; m < MP; ++m) s += qs[m] * ctxb[m * DHD + tid];
    const int bb = bh / NH, hh = bh % NH;
    O[((size_t)(bb * SEQL + n)) * DIM + hh * DHD + tid] = (_Float16)(s * dinv);
  }
}

// ============================================================================
// Elementwise / small kernels
// ============================================================================
__global__ void embed0_kernel(const float* __restrict__ mag, const float* __restrict__ syn,
                              const float* __restrict__ se, float* __restrict__ E) {
  const size_t total = (size_t)BS * SEQL * DIM;
  for (size_t idx = (size_t)blockIdx.x * blockDim.x + threadIdx.x; idx < total;
       idx += (size_t)gridDim.x * blockDim.x) {
    int d = (int)(idx % DIM);
    size_t rt = idx / DIM;
    int t = (int)(rt % SEQL);
    int b = (int)(rt / SEQL);
    int l = t / 192, c = t % 192;
    float tok = (c < 128) ? mag[((size_t)b * 8 + l) * 128 + c]
                          : syn[((size_t)b * 8 + l) * 64 + (c - 128)];
    E[idx] = tok * se[(size_t)t * DIM + d];
  }
}

__global__ void embedT_kernel(const float* __restrict__ magT, const float* __restrict__ synT,
                              const float* __restrict__ se, float* __restrict__ E) {
  const size_t total = (size_t)BS * SEQL * DIM;
  for (size_t idx = (size_t)blockIdx.x * blockDim.x + threadIdx.x; idx < total;
       idx += (size_t)gridDim.x * blockDim.x) {
    int d = (int)(idx % DIM);
    size_t rt = idx / DIM;
    int t = (int)(rt % SEQL);
    int b = (int)(rt / SEQL);
    int r = t / 8, l = t % 8;   // t = r*8 + l
    float tok = (r < 128) ? magT[((size_t)b * 128 + r) * 8 + l]
                          : synT[((size_t)b * 64 + (r - 128)) * 8 + l];
    E[idx] = tok * se[(size_t)t * DIM + d];
  }
}

__global__ void combine_kernel(const float* __restrict__ Y1, const float* __restrict__ Y2,
                               float* __restrict__ X) {
  const size_t total = (size_t)NROWS * DIM;
  for (size_t idx = (size_t)blockIdx.x * blockDim.x + threadIdx.x; idx < total;
       idx += (size_t)gridDim.x * blockDim.x)
    X[idx] = 0.5f * (Y1[idx] + Y2[idx]);
}

__global__ void merge_kernel(float* __restrict__ E, const float* __restrict__ ET) {
  const size_t total = (size_t)NROWS * DIM;
  for (size_t idx = (size_t)blockIdx.x * blockDim.x + threadIdx.x; idx < total;
       idx += (size_t)gridDim.x * blockDim.x) {
    int d = (int)(idx % DIM);
    size_t rt = idx / DIM;
    int t = (int)(rt % SEQL);
    int b = (int)(rt / SEQL);
    int l = t / 192, r = t % 192;   // t = l*192 + r
    int tT = r * 8 + l;
    E[idx] += ET[((size_t)b * SEQL + tT) * DIM + d];
  }
}

__global__ __launch_bounds__(256)
void zfinal_kernel(const float* __restrict__ E, const float* __restrict__ fw,
                   const float* __restrict__ fb, float* __restrict__ Z) {
  __shared__ float red[256];
  const int tid = threadIdx.x;
  const size_t base = (size_t)blockIdx.x * DIM;
  red[tid] = E[base + tid] * fw[tid];
  __syncthreads();
  for (int s = 128; s > 0; s >>= 1) {
    if (tid < s) red[tid] += red[tid + s];
    __syncthreads();
  }
  if (tid == 0) Z[blockIdx.x] = red[0] + fb[0];
}

__global__ __launch_bounds__(256)
void out_kernel(const float* __restrict__ Z, const float* __restrict__ OW,
                const float* __restrict__ OB, float* __restrict__ OUT) {
  const int j = blockIdx.x * 256 + threadIdx.x;   // 0..1023
  const int b = blockIdx.y;
  float s = OB[j];
  const float* zb = Z + (size_t)b * SEQL;
  for (int t = 0; t < SEQL; ++t) s += zb[t] * OW[(size_t)t * 1024 + j];
  OUT[(size_t)b * 1024 + j] = s;
}

__global__ void fill_u32_kernel(unsigned* p, unsigned v, int n) {
  int i = blockIdx.x * blockDim.x + threadIdx.x;
  if (i < n) p[i] = v;
}

// ============================================================================
// Host orchestration
// ============================================================================
extern "C" void kernel_launch(void* const* d_in, const int* in_sizes, int n_in,
                              void* d_out, int out_size, void* d_ws, size_t ws_size,
                              hipStream_t stream) {
  (void)in_sizes; (void)n_in; (void)out_size; (void)ws_size;
  const float* magnitude   = (const float*)d_in[0];
  const float* syndrome    = (const float*)d_in[1];
  const float* magnitude_T = (const float*)d_in[2];
  const float* syndrome_T  = (const float*)d_in[3];
  const float* src_embed   = (const float*)d_in[4];
  const float* src_embed_T = (const float*)d_in[5];
  const float* ln1_s = (const float*)d_in[6];
  const float* ln1_b = (const float*)d_in[7];
  const float* Wq = (const float*)d_in[8];
  const float* bq = (const float*)d_in[9];
  const float* Wk = (const float*)d_in[10];
  const float* bk = (const float*)d_in[11];
  const float* Wv = (const float*)d_in[12];
  const float* bv = (const float*)d_in[13];
  const float* Wo = (const float*)d_in[14];
  const float* bo = (const float*)d_in[15];
  const float* proj = (const float*)d_in[16];
  const float* ln2_s = (const float*)d_in[17];
  const float* ln2_b = (const float*)d_in[18];
  const float* W1 = (const float*)d_in[19];
  const float* b1 = (const float*)d_in[20];
  const float* W2 = (const float*)d_in[21];
  const float* b2 = (const float*)d_in[22];
  const float* final_w = (const float*)d_in[23];
  const float* final_b = (const float*)d_in[24];
  const float* out_w = (const float*)d_in[25];
  const float* out_b = (const float*)d_in[26];
  float* outp = (float*)d_out;

  float* ws = (float*)d_ws;
  size_t off = 0;
  auto alloc = [&](size_t n) { float* p = ws + off; off += (n + 15) & ~(size_t)15; return p; };
  auto allocH = [&](size_t n) { return (_Float16*)alloc((n + 1) / 2); };

  const size_t RD = (size_t)NROWS * DIM;
  float* embA = alloc(RD);
  float* embT = alloc(RD);
  float* qbuf = alloc(RD);
  float* kbuf = alloc(RD);
  float* vbuf = alloc(RD);
  _Float16* hbuf16 = allocH(RD);
  _Float16* obuf16 = allocH(RD);
  _Float16* ffa16  = allocH((size_t)NROWS * 2 * DIM);
  const size_t PH = (size_t)BS * NH * SEQL * MP;
  float* qp   = alloc(PH);
  float* kp   = alloc(PH);
  float* ksum = alloc((size_t)BS * NH * MP);
  unsigned* kmax = (unsigned*)alloc(BS * NH);
  float* ctx  = alloc((size_t)BS * NH * MP * DHD);
  float* zb   = alloc(NROWS);
  // f16 transposed weights [lid][N][K]
  const size_t WSQ = (size_t)12 * DIM * DIM;          // per Wq/Wk/Wv/Wo set
  const size_t WSF = (size_t)12 * DIM * 2 * DIM;      // per W1/W2 set
  _Float16* WqT = allocH(WSQ);
  _Float16* WkT = allocH(WSQ);
  _Float16* WvT = allocH(WSQ);
  _Float16* WoT = allocH(WSQ);
  _Float16* W1T = allocH(WSF);
  _Float16* W2T = allocH(WSF);
  // safe aliases (stream-serialized dependencies)
  float* y1 = kbuf;   // written after phi_k1 consumed k
  float* y2 = vbuf;   // written after ctx consumed v

  // one-time (per launch) weight convert + transpose to f16
  convT_kernel<<<2048, 256, 0, stream>>>(Wq, WqT, DIM, DIM, 12);
  convT_kernel<<<2048, 256, 0, stream>>>(Wk, WkT, DIM, DIM, 12);
  convT_kernel<<<2048, 256, 0, stream>>>(Wv, WvT, DIM, DIM, 12);
  convT_kernel<<<2048, 256, 0, stream>>>(Wo, WoT, DIM, DIM, 12);
  convT_kernel<<<2048, 256, 0, stream>>>(W1, W1T, DIM, 2 * DIM, 12);
  convT_kernel<<<2048, 256, 0, stream>>>(W2, W2T, 2 * DIM, DIM, 12);

  const int EW = 8192;
  embed0_kernel<<<EW, 256, 0, stream>>>(magnitude, syndrome, src_embed, embA);
  embedT_kernel<<<EW, 256, 0, stream>>>(magnitude_T, syndrome_T, src_embed_T, embT);

  const dim3 g256(4, NROWS / 64), g512(8, NROWS / 64);
  const int BHN = BS * NH * SEQL;

  for (int s = 0; s < 2; ++s) {
    float* x = (s == 0) ? embA : embT;
    for (int l = 0; l < 6; ++l) {
      const size_t lid = (size_t)s * 6 + l;
      const float* pP = proj + lid * MF * DHD;
      const size_t oQ = lid * (size_t)DIM * DIM;
      const size_t oF = lid * (size_t)DIM * 2 * DIM;
      ln_kernel<<<NROWS, 256, 0, stream>>>(x, ln1_s + lid * DIM, ln1_b + lid * DIM, hbuf16);
      gemm_f16_kernel<0, 0><<<g256, 128, 0, stream>>>(hbuf16, WqT + oQ, bq + lid * DIM, nullptr, qbuf, DIM, DIM);
      gemm_f16_kernel<0, 0><<<g256, 128, 0, stream>>>(hbuf16, WkT + oQ, bk + lid * DIM, nullptr, kbuf, DIM, DIM);
      gemm_f16_kernel<0, 0><<<g256, 128, 0, stream>>>(hbuf16, WvT + oQ, bv + lid * DIM, nullptr, vbuf, DIM, DIM);
      phi_q_kernel<<<BHN, 128, 0, stream>>>(qbuf, pP, qp);
      fill_u32_kernel<<<1, 256, 0, stream>>>(kmax, 0u, BS * NH);
      phi_k1_kernel<<<BHN, 128, 0, stream>>>(kbuf, pP, kp, kmax);
      phi_k2_kernel<<<EW, 256, 0, stream>>>(kp, kmax);
      ksum_kernel<<<BS * NH, 128, 0, stream>>>(kp, ksum);
      ctx_kernel<<<BS * NH, 128, 0, stream>>>(kp, vbuf, ctx);
      attn_o_kernel<<<BHN, 128, 0, stream>>>(qp, ctx, ksum, obuf16);
      gemm_f16_kernel<2, 0><<<g256, 128, 0, stream>>>(obuf16, WoT + oQ, bo + lid * DIM, x, y1, DIM, DIM);
      ln_kernel<<<NROWS, 256, 0, stream>>>(y1, ln2_s + lid * DIM, ln2_b + lid * DIM, hbuf16);
      gemm_f16_kernel<1, 1><<<g512, 128, 0, stream>>>(hbuf16, W1T + oF, b1 + lid * 2 * DIM, nullptr, ffa16, DIM, 2 * DIM);
      gemm_f16_kernel<2, 0><<<g256, 128, 0, stream>>>(ffa16, W2T + oF, b2 + lid * DIM, x, y2, 2 * DIM, DIM);
      combine_kernel<<<EW, 256, 0, stream>>>(y1, y2, x);
    }
  }
  merge_kernel<<<EW, 256, 0, stream>>>(embA, embT);
  zfinal_kernel<<<NROWS, 256, 0, stream>>>(embA, final_w, final_b, zb);
  out_kernel<<<dim3(4, BS), 256, 0, stream>>>(zb, out_w, out_b, outp);
}
